// AllegroLayer_62362925137935
// MI455X (gfx1250) — matrix operand order repaired and checked by reading, not compiled, not run on hardware
//
#include <hip/hip_runtime.h>
#include <hip/hip_bf16.h>
#include <math.h>

#define N_NODES  25000
#define N_EDGES  400000
#define FEAT     128
#define NCH      32
#define FEAT_OUT 128
#define NOUT     32

typedef __attribute__((ext_vector_type(16))) __bf16 v16bf;
typedef __attribute__((ext_vector_type(8)))  __bf16 v8bf;
typedef __attribute__((ext_vector_type(8)))  float  v8f;

static __device__ __forceinline__ v16bf joinbf(v8bf lo, v8bf hi) {
  return __builtin_shufflevector(lo, hi, 0,1,2,3,4,5,6,7,8,9,10,11,12,13,14,15);
}

static __device__ __forceinline__ v8f wmma_bf16(v16bf a, v16bf b, v8f c) {
  // D(f32 16x16) = A(16x32 bf16) x B(32x16 bf16) + C
  return __builtin_amdgcn_wmma_f32_16x16x32_bf16(false, a, false, b, (short)0, c, false, false);
}

// A fragment (16x32, 16-bit) from f32 row-major source.
// lane: M = lane&15, half = lane>>4. e<8 -> K = kbase+half*8+e ; e>=8 -> K = kbase+16+half*8+(e-8)
static __device__ __forceinline__ v16bf load_a_f32(const float* __restrict__ rowptr, int kbase, int half) {
  const float4* p0 = (const float4*)(rowptr + kbase + half * 8);
  const float4* p1 = (const float4*)(rowptr + kbase + 16 + half * 8);
  float4 q0 = p0[0], q1 = p0[1];
  float4 q2 = p1[0], q3 = p1[1];
  v16bf a;
  a[0]=(__bf16)q0.x;  a[1]=(__bf16)q0.y;  a[2]=(__bf16)q0.z;  a[3]=(__bf16)q0.w;
  a[4]=(__bf16)q1.x;  a[5]=(__bf16)q1.y;  a[6]=(__bf16)q1.z;  a[7]=(__bf16)q1.w;
  a[8]=(__bf16)q2.x;  a[9]=(__bf16)q2.y;  a[10]=(__bf16)q2.z; a[11]=(__bf16)q2.w;
  a[12]=(__bf16)q3.x; a[13]=(__bf16)q3.y; a[14]=(__bf16)q3.z; a[15]=(__bf16)q3.w;
  return a;
}

// Same A fragment but from a bf16 row (global or LDS) -> two b128 loads
static __device__ __forceinline__ v16bf load_a_bf16(const __bf16* rowptr, int kbase, int half) {
  const v8bf* p0 = (const v8bf*)(rowptr + kbase + half * 8);
  const v8bf* p1 = (const v8bf*)(rowptr + kbase + 16 + half * 8);
  return joinbf(*p0, *p1);
}

// B fragment from pre-packed fragment layout: lane reads 16 contiguous halves.
static __device__ __forceinline__ v16bf load_wfrag(const __bf16* __restrict__ wf, int f, int lane) {
  const v8bf* p = (const v8bf*)(wf + (size_t)(f * 32 + lane) * 16);
  return joinbf(p[0], p[1]);
}

// ---------------- kernel 0: zero agg + pack weights to bf16 B-fragment layout ----------------
// B fragment layout for W (K x N), NT = N/16 tiles, KC = K/32 chunks, f = kk*NT+nt:
//   dst[(f*32 + lane)*16 + e] = W[kk*32 + (e<8 ? (lane>>4)*8 + e : 16 + (lane>>4)*8 + (e-8))][nt*16 + (lane&15)]
static __device__ __forceinline__ void prep_one(const float* __restrict__ W, int N, int NT,
                                                __bf16* __restrict__ dst, int t) {
  int f   = t >> 9;        // /512
  int rem = t & 511;
  int l   = rem >> 4;
  int e   = rem & 15;
  int kk  = f / NT, nt = f % NT;
  int krow = kk * 32 + (e < 8 ? (l >> 4) * 8 + e : 16 + (l >> 4) * 8 + (e - 8));
  int col  = nt * 16 + (l & 15);
  dst[t] = (__bf16)W[(size_t)krow * N + col];
}

__global__ __launch_bounds__(256) void k_prep(
    const float* __restrict__ W1,  const float* __restrict__ W2a,
    const float* __restrict__ W2b, const float* __restrict__ W2c,
    const float* __restrict__ Wlin,
    __bf16* __restrict__ w1f,  __bf16* __restrict__ w2af,
    __bf16* __restrict__ w2bf, __bf16* __restrict__ w2cf,
    __bf16* __restrict__ wlinf, float* __restrict__ agg) {
  int tid = blockIdx.x * 256 + threadIdx.x;
  if (tid < N_NODES * NCH * 4) agg[tid] = 0.f;   // 3.2M floats
  int t = tid;
  if (t < 4096)  { prep_one(W1,   32, 2, w1f,  t); return; }  // 128x32
  t -= 4096;
  if (t < 24576) { prep_one(W2a, 128, 8, w2af, t); return; }  // 192x128
  t -= 24576;
  if (t < 16384) { prep_one(W2b, 128, 8, w2bf, t); return; }  // 128x128
  t -= 16384;
  if (t < 16384) { prep_one(W2c, 128, 8, w2cf, t); return; }  // 128x128
  t -= 16384;
  if (t < 2048)  { prep_one(Wlin, 32, 2, wlinf, t); return; } // 64x32
}

// ---------------- kernel 1: w = x@W1/sqrt(128); atomic scatter of w*Y into agg ----------------
__global__ __launch_bounds__(256) void k_w_scatter(
    const float* __restrict__ x, const float* __restrict__ Y,
    const int* __restrict__ edge_src, const __bf16* __restrict__ w1f,
    float* __restrict__ agg) {
  int lane = threadIdx.x & 31;
  int tile = (blockIdx.x * 256 + threadIdx.x) >> 5;  // 25000 tiles exactly
  int ebase = tile * 16;
  int row = lane & 15, half = lane >> 4;
  const float* xrow = x + (size_t)(ebase + row) * FEAT;
  v16bf a[4];
#pragma unroll
  for (int kk = 0; kk < 4; ++kk) a[kk] = load_a_f32(xrow, kk * 32, half);
  v8f acc0 = {}, acc1 = {};
#pragma unroll
  for (int kk = 0; kk < 4; ++kk) {
    acc0 = wmma_bf16(a[kk], load_wfrag(w1f, kk * 2 + 0, lane), acc0);
    acc1 = wmma_bf16(a[kk], load_wfrag(w1f, kk * 2 + 1, lane), acc1);
  }
  const float sc = 0.08838834764831845f; // 1/sqrt(128)
  const float4* Yv = (const float4*)Y;
  int ch0 = row, ch1 = 16 + row;
#pragma unroll
  for (int j = 0; j < 8; ++j) {
    int edge = ebase + half * 8 + j;       // C layout: M = half*8 + j
    int src = edge_src[edge];
    float4 y = Yv[edge];
    float w0 = acc0[j] * sc, w1 = acc1[j] * sc;
    float* p0 = agg + (size_t)src * (NCH * 4) + ch0 * 4;
    float* p1 = agg + (size_t)src * (NCH * 4) + ch1 * 4;
    atomicAdd(p0 + 0, w0 * y.x); atomicAdd(p0 + 1, w0 * y.y);
    atomicAdd(p0 + 2, w0 * y.z); atomicAdd(p0 + 3, w0 * y.w);
    atomicAdd(p1 + 0, w1 * y.x); atomicAdd(p1 + 1, w1 * y.y);
    atomicAdd(p1 + 2, w1 * y.z); atomicAdd(p1 + 3, w1 * y.w);
  }
}

// ---------------- kernel 2 (fused): gather + scalars + envelope + V_out WMMA + 3-layer MLP ----------------
__global__ __launch_bounds__(128) void k_edge(
    const float* __restrict__ x, const float* __restrict__ V,
    const float* __restrict__ r, const int* __restrict__ edge_src,
    const float* __restrict__ agg,
    const __bf16* __restrict__ w2af, const __bf16* __restrict__ w2bf,
    const __bf16* __restrict__ w2cf, const __bf16* __restrict__ wlinf,
    float* __restrict__ xout, float* __restrict__ vout) {
  __shared__ __attribute__((aligned(16))) __bf16 vvb[4][3][16][64];   // Vv components, [wave][i][row][u]
  __shared__ __attribute__((aligned(16))) __bf16 scalb[4][16][64];    // scalars (bf16)
  __shared__ __attribute__((aligned(16))) __bf16 hb[4][16][128];      // MLP activation staging
  __shared__ float envb[4][16];

  int w = threadIdx.x >> 5, lane = threadIdx.x & 31;
  int tile = blockIdx.x * 4 + w;        // 25000 tiles exactly
  int ebase = tile * 16;
  int row = lane & 15, half = lane >> 4;
  int edge = ebase + row;
  int src = edge_src[edge];

  const float4* aggv = (const float4*)agg;
  const float4* Vq = (const float4*)V;
#pragma unroll
  for (int cc = 0; cc < 16; ++cc) {
    int c = half * 16 + cc;
    float4 a = aggv[(size_t)src * NCH + c];   // agg / sqrt(16) below
    float4 b = Vq[(size_t)edge * NCH + c];
    float as = a.x * 0.25f, av0 = a.y * 0.25f, av1 = a.z * 0.25f, av2 = a.w * 0.25f;
    float s1 = as * b.x;
    float s2 = (av0 * b.y + av1 * b.z + av2 * b.w) * 0.5773502691896258f; // 1/sqrt(3)
    scalb[w][row][2 * c]     = (__bf16)s1;
    scalb[w][row][2 * c + 1] = (__bf16)s2;
    // v1 = a_s * b_v (u=2c), v2 = a_v * b_s (u=2c+1)
    vvb[w][0][row][2 * c]     = (__bf16)(as * b.y);
    vvb[w][1][row][2 * c]     = (__bf16)(as * b.z);
    vvb[w][2][row][2 * c]     = (__bf16)(as * b.w);
    vvb[w][0][row][2 * c + 1] = (__bf16)(av0 * b.x);
    vvb[w][1][row][2 * c + 1] = (__bf16)(av1 * b.x);
    vvb[w][2][row][2 * c + 1] = (__bf16)(av2 * b.x);
  }
  if (half == 0) {  // envelope(d), p=6: 1 - 28 d^6 + 48 d^7 - 21 d^8
    float rx = r[(size_t)edge * 3], ry = r[(size_t)edge * 3 + 1], rz = r[(size_t)edge * 3 + 2];
    float x2 = rx * rx + ry * ry + rz * rz;
    float d = sqrtf(x2 == 0.f ? 1.f : x2);
    float d2 = d * d, d6 = d2 * d2 * d2;
    float poly = 1.f + d6 * (-28.f + d * (48.f - 21.f * d));
    envb[w][row] = (d < 1.f) ? poly : 0.f;
  }
  __syncthreads();

  // ---- V_out[e,v,i] = sum_u Vv[e,u,i] * Wlin[u,v] / 8 ----
#pragma unroll
  for (int i = 0; i < 3; ++i) {
    v16bf A0 = load_a_bf16(&vvb[w][i][row][0], 0, half);
    v16bf A1 = load_a_bf16(&vvb[w][i][row][0], 32, half);
#pragma unroll
    for (int nt = 0; nt < 2; ++nt) {
      v8f acc = {};
      acc = wmma_bf16(A0, load_wfrag(wlinf, 0 * 2 + nt, lane), acc);
      acc = wmma_bf16(A1, load_wfrag(wlinf, 1 * 2 + nt, lane), acc);
#pragma unroll
      for (int j = 0; j < 8; ++j) {
        int e2 = ebase + half * 8 + j;
        int v = nt * 16 + row;
        vout[(size_t)e2 * (NCH * 3) + v * 3 + i] = acc[j] * 0.125f;
      }
    }
  }

  // ---- MLP: h = concat(x, scalars) -> W2a -> silu -> W2b -> silu -> W2c -> env scale ----
  const float* xrow = x + (size_t)edge * FEAT;
  v16bf a6[6];
#pragma unroll
  for (int kk = 0; kk < 4; ++kk) a6[kk] = load_a_f32(xrow, kk * 32, half);
  a6[4] = load_a_bf16(&scalb[w][row][0], 0, half);
  a6[5] = load_a_bf16(&scalb[w][row][0], 32, half);

  const float sA = 0.07216878364870323f; // 1/sqrt(192)
#pragma unroll
  for (int nt = 0; nt < 8; ++nt) {
    v8f acc = {};
#pragma unroll
    for (int kk = 0; kk < 6; ++kk)
      acc = wmma_bf16(a6[kk], load_wfrag(w2af, kk * 8 + nt, lane), acc);
#pragma unroll
    for (int j = 0; j < 8; ++j) {
      float z = acc[j] * sA;
      float h = z / (1.f + __expf(-z));   // silu
      hb[w][half * 8 + j][nt * 16 + row] = (__bf16)h;
    }
  }
  __syncthreads();

  const float sB = 0.08838834764831845f; // 1/sqrt(128)
  v16bf b4[4];
#pragma unroll
  for (int kk = 0; kk < 4; ++kk) b4[kk] = load_a_bf16(&hb[w][row][0], kk * 32, half);
#pragma unroll
  for (int nt = 0; nt < 8; ++nt) {
    v8f acc = {};
#pragma unroll
    for (int kk = 0; kk < 4; ++kk)
      acc = wmma_bf16(b4[kk], load_wfrag(w2bf, kk * 8 + nt, lane), acc);
#pragma unroll
    for (int j = 0; j < 8; ++j) {
      float z = acc[j] * sB;
      float h = z / (1.f + __expf(-z));
      hb[w][half * 8 + j][nt * 16 + row] = (__bf16)h;
    }
  }
  __syncthreads();

  v16bf c4[4];
#pragma unroll
  for (int kk = 0; kk < 4; ++kk) c4[kk] = load_a_bf16(&hb[w][row][0], kk * 32, half);
  float ev[8];
#pragma unroll
  for (int j = 0; j < 8; ++j) ev[j] = envb[w][half * 8 + j];
#pragma unroll
  for (int nt = 0; nt < 8; ++nt) {
    v8f acc = {};
#pragma unroll
    for (int kk = 0; kk < 4; ++kk)
      acc = wmma_bf16(c4[kk], load_wfrag(w2cf, kk * 8 + nt, lane), acc);
#pragma unroll
    for (int j = 0; j < 8; ++j) {
      int e2 = ebase + half * 8 + j;
      xout[(size_t)e2 * FEAT_OUT + nt * 16 + row] = acc[j] * sB * ev[j];
    }
  }
}

extern "C" void kernel_launch(void* const* d_in, const int* in_sizes, int n_in,
                              void* d_out, int out_size, void* d_ws, size_t ws_size,
                              hipStream_t stream) {
  const float* x    = (const float*)d_in[0];
  const float* V    = (const float*)d_in[1];
  const float* r    = (const float*)d_in[2];
  const float* Y    = (const float*)d_in[3];
  const int*   esrc = (const int*)d_in[4];
  const float* W1   = (const float*)d_in[5];
  const float* W2a  = (const float*)d_in[6];
  const float* W2b  = (const float*)d_in[7];
  const float* W2c  = (const float*)d_in[8];
  const float* Wlin = (const float*)d_in[9];

  // workspace layout (bytes), all 256-aligned:
  //   agg   : 25000*128*4 = 12,800,000
  //   w1f   : 8192   w2af: 49152   w2bf: 32768   w2cf: 32768   wlinf: 4096
  char* ws = (char*)d_ws;
  float*  agg   = (float*)(ws + 0);
  __bf16* w1f   = (__bf16*)(ws + 12800000);
  __bf16* w2af  = (__bf16*)(ws + 12808192);
  __bf16* w2bf  = (__bf16*)(ws + 12857344);
  __bf16* w2cf  = (__bf16*)(ws + 12890112);
  __bf16* wlinf = (__bf16*)(ws + 12922880);

  float* xout = (float*)d_out;
  float* vout = xout + (size_t)N_EDGES * FEAT_OUT;  // outputs concatenated flat

  (void)in_sizes; (void)n_in; (void)out_size; (void)ws_size;

  k_prep<<<12500, 256, 0, stream>>>(W1, W2a, W2b, W2c, Wlin,
                                    w1f, w2af, w2bf, w2cf, wlinf, agg);
  k_w_scatter<<<3125, 256, 0, stream>>>(x, Y, esrc, w1f, agg);
  k_edge<<<6250, 128, 0, stream>>>(x, V, r, esrc, agg,
                                   w2af, w2bf, w2cf, wlinf, xout, vout);
}
